// DepthPredictorMultiView360_67430986547504
// MI455X (gfx1250) — matrix operand order repaired and checked by reading, not compiled, not run on hardware
//
#include <hip/hip_runtime.h>
#include <hip/hip_bf16.h>
#include <math.h>

typedef _Float16 v16h __attribute__((ext_vector_type(16)));
typedef _Float16 h8   __attribute__((ext_vector_type(8)));
typedef _Float16 h4   __attribute__((ext_vector_type(4)));
typedef float    v8f  __attribute__((ext_vector_type(8)));

#define NVIEW 2
#define C 128
#define H 64
#define W 128
#define D 32
#define TWO_PI 6.283185307179586f
#define PI_F   3.14159265358979323846f

static constexpr size_t T_BYTES = (size_t)NVIEW * H * W * C * sizeof(_Float16);

// ---------------- Kernel 1: NCHW f32 -> NHWC f16 transpose (per view) ----------------
__global__ void k_transpose(const float* __restrict__ feat, _Float16* __restrict__ T) {
  const int h = blockIdx.x;   // 0..63
  const int n = blockIdx.y;   // view 0..1
  __shared__ float tile[32][33];
  const int tw = threadIdx.x & 31;
  const int tc = threadIdx.x >> 5;   // 0..7
  for (int c0 = 0; c0 < C; c0 += 32) {
    for (int w0 = 0; w0 < W; w0 += 32) {
#pragma unroll
      for (int i = 0; i < 32; i += 8)
        tile[tc + i][tw] = feat[(((size_t)n * C + (c0 + tc + i)) * H + h) * W + (w0 + tw)];
      __syncthreads();
#pragma unroll
      for (int i = 0; i < 32; i += 8)
        T[(((size_t)n * H + h) * W + (w0 + tc + i)) * C + (c0 + tw)] = (_Float16)tile[tw][tc + i];
      __syncthreads();
    }
  }
}

// ---------------- Kernel 2: poses (rigid-inverse closed form) + inverse-depth table ----------------
__global__ void k_setup(const float* __restrict__ ext, const float* __restrict__ nearv,
                        const float* __restrict__ farv, float* __restrict__ P) {
  if (threadIdx.x == 0) {
    float R0[3][3], R1[3][3], t0[3], t1[3];
    for (int i = 0; i < 3; i++) {
      for (int j = 0; j < 3; j++) { R0[i][j] = ext[0 * 16 + i * 4 + j]; R1[i][j] = ext[1 * 16 + i * 4 + j]; }
      t0[i] = ext[0 * 16 + i * 4 + 3]; t1[i] = ext[1 * 16 + i * 4 + 3];
    }
    float Rp[3][3], tp[3];
    for (int i = 0; i < 3; i++)
      for (int j = 0; j < 3; j++) {
        float s = 0.f;
        for (int k = 0; k < 3; k++) s += R1[k][i] * R0[k][j];
        Rp[i][j] = s;
      }
    for (int i = 0; i < 3; i++) {
      float s = 0.f;
      for (int k = 0; k < 3; k++) s += R1[k][i] * (t0[k] - t1[k]);
      tp[i] = s;
    }
    for (int i = 0; i < 3; i++)
      for (int j = 0; j < 3; j++) { P[i * 3 + j] = Rp[i][j]; P[12 + i * 3 + j] = Rp[j][i]; }
    for (int i = 0; i < 3; i++) {
      P[9 + i] = tp[i];
      float s = 0.f;
      for (int k = 0; k < 3; k++) s += Rp[k][i] * tp[k];
      P[12 + 9 + i] = -s;
    }
  }
  const int k = threadIdx.x;
  if (k < D) {
    for (int n = 0; n < NVIEW; n++) {
      float mind = 1.0f / farv[n];
      float maxd = 1.0f / nearv[n];
      float lin = (float)k / (float)(D - 1);
      P[24 + n * D + k] = 1.0f / (mind + lin * (maxd - mind));
    }
  }
}

// ---------------- fragment loaders (CDNA5 16-bit WMMA layouts) ----------------
// A 16x32 f16: lane<16 -> row=lane, K = 0..7 (v0-3) and 16..23 (v4-7); lane>=16 -> K +8
__device__ inline v16h fragA_from_lds(const _Float16* rowbase, int kbase, int lane) {
  const int kb = (lane >> 4) << 3;
  h8 lo = *(const h8*)(rowbase + kbase + kb);
  h8 hi = *(const h8*)(rowbase + kbase + 16 + kb);
  v16h r;
#pragma unroll
  for (int i = 0; i < 8; ++i) { r[i] = lo[i]; r[i + 8] = hi[i]; }
  return r;
}
// B 32x16 f16: lane<16 -> col=lane, K = 0..7 (v0-3) and 16..23 (v4-7); lane>=16 -> col=lane-16, K +8
__device__ inline v16h fragB_from_lds(const _Float16* colbase, int kbase, int lane) {
  const _Float16* q = colbase + kbase + ((lane >> 4) << 3);
  h8 lo = *(const h8*)q;
  h8 hi = *(const h8*)(q + 16);
  v16h r;
#pragma unroll
  for (int i = 0; i < 8; ++i) { r[i] = lo[i]; r[i + 8] = hi[i]; }
  return r;
}

// wave-local LDS ordering: HW keeps same-wave DS ops in order (DScnt in-order);
// we only need to pin the compiler's schedule.
__device__ inline void wave_lds_order() {
  __builtin_amdgcn_fence(__ATOMIC_ACQ_REL, "wavefront");
  __builtin_amdgcn_wave_barrier();
}

// ---------------- Kernel 3: warp + gather + WMMA correlation ----------------
__global__ __launch_bounds__(256) void k_corr(const _Float16* __restrict__ T,
                                              const float* __restrict__ P,
                                              float* __restrict__ out) {
  const int wt = blockIdx.x;   // w-tile 0..7
  const int h  = blockIdx.y;   // 0..63
  const int n  = blockIdx.z;   // 0..1
  const int w0 = wt * 16;
  const int lane = threadIdx.x & 31;
  const int wave = threadIdx.x >> 5;

  __shared__ __align__(16) _Float16 sB[16 * C];       // feat0 pixel vectors (B matrix)
  __shared__ __align__(16) _Float16 sA[8][16 * C];    // per-wave warped vectors (A matrix)

  // stage feat0 (view n) 16 pixel vectors (16*C*2 = 4096B = 256 threads x 16B)
  {
    const uint4* s4 = (const uint4*)(T + (((size_t)n * H + h) * W + w0) * C);
    uint4* d4 = (uint4*)sB;
    d4[threadIdx.x] = s4[threadIdx.x];
  }
  __syncthreads();   // block-wide: every wave reads all of sB

  const int p = lane & 15;  // pixel within tile (A-row / B-col index for this lane)

  v16h bfrag[4];
#pragma unroll
  for (int kk = 0; kk < 4; ++kk)
    bfrag[kk] = fragB_from_lds(sB + (size_t)p * C, kk * 32, lane);

  const float r00 = P[n * 12 + 0], r01 = P[n * 12 + 1], r02 = P[n * 12 + 2];
  const float r10 = P[n * 12 + 3], r11 = P[n * 12 + 4], r12 = P[n * 12 + 5];
  const float r20 = P[n * 12 + 6], r21 = P[n * 12 + 7], r22 = P[n * 12 + 8];
  const float tx = P[n * 12 + 9], ty = P[n * 12 + 10], tz = P[n * 12 + 11];

  const _Float16* F1 = T + (size_t)(1 - n) * H * W * C;   // feat1 = other view
  _Float16* sAw = sA[wave];

  const float theta = ((float)(w0 + p) + 0.5f) * (TWO_PI / W);
  const float phi   = ((float)h + 0.5f) * (PI_F / H);
  float sphi, cphi, sth, cth;
  __sincosf(phi, &sphi, &cphi);
  __sincosf(theta, &sth, &cth);
  const float dx = sphi * sth, dy = cphi, dz = sphi * cth;

  for (int dep = wave; dep < D; dep += 8) {     // 4 iterations, waves fully decoupled
    const float z = P[24 + n * D + dep];
    // --- warp this lane's pixel to source view ---
    const float px = z * dx, py = z * dy, pz = z * dz;
    const float wxp = r00 * px + r01 * py + r02 * pz + tx;
    const float wyp = r10 * px + r11 * py + r12 * pz + ty;
    const float wzp = r20 * px + r21 * py + r22 * pz + tz;
    float rr = fmaxf(sqrtf(wxp * wxp + wyp * wyp + wzp * wzp), 0.001f);
    const float cy = fminf(fmaxf(wyp / rr, -1.0f), 1.0f);
    const float phiw = acosf(cy);
    float thw = atan2f(wxp, wzp);
    if (thw < 0.0f) thw += TWO_PI;
    const float u = thw * (1.0f / PI_F) - 1.0f;
    const float v = phiw * (2.0f / PI_F) - 1.0f;
    const float ix = (u + 1.0f) * 0.5f * (float)(W - 1);
    const float iy = (v + 1.0f) * 0.5f * (float)(H - 1);
    const float x0f = floorf(ix), y0f = floorf(iy);
    const float fx = ix - x0f, fy = iy - y0f;
    float w00 = (1.0f - fx) * (1.0f - fy);
    float w10 = fx * (1.0f - fy);
    float w01 = (1.0f - fx) * fy;
    float w11 = fx * fy;
    const int x0 = (int)x0f, y0i = (int)y0f;
    const int x1 = x0 + 1, y1 = y0i + 1;
    if (x0 < 0 || x0 > W - 1) { w00 = 0.0f; w01 = 0.0f; }
    if (x1 < 0 || x1 > W - 1) { w10 = 0.0f; w11 = 0.0f; }
    if (y0i < 0 || y0i > H - 1) { w00 = 0.0f; w10 = 0.0f; }
    if (y1 < 0 || y1 > H - 1) { w01 = 0.0f; w11 = 0.0f; }
    const int x0c = min(max(x0, 0), W - 1), x1c = min(max(x1, 0), W - 1);
    const int y0c = min(max(y0i, 0), H - 1), y1c = min(max(y1, 0), H - 1);
    const int o00 = (y0c * W + x0c) * C;
    const int o10 = (y0c * W + x1c) * C;
    const int o01 = (y1c * W + x0c) * C;
    const int o11 = (y1c * W + x1c) * C;

    // --- cooperative gather + bilinear blend: full unroll => literal-lane shfl
    //     (v_readlane -> wave-uniform SGPR weights/addresses) ---
#pragma unroll
    for (int q = 0; q < 16; ++q) {
      const int a00 = __shfl(o00, q, 32);
      const int a10 = __shfl(o10, q, 32);
      const int a01 = __shfl(o01, q, 32);
      const int a11 = __shfl(o11, q, 32);
      const float f00 = __shfl(w00, q, 32);
      const float f10 = __shfl(w10, q, 32);
      const float f01 = __shfl(w01, q, 32);
      const float f11 = __shfl(w11, q, 32);
      const int co = lane * 4;
      const h4 t00 = *(const h4*)(F1 + a00 + co);
      const h4 t10 = *(const h4*)(F1 + a10 + co);
      const h4 t01 = *(const h4*)(F1 + a01 + co);
      const h4 t11 = *(const h4*)(F1 + a11 + co);
      const _Float16 g00 = (_Float16)f00, g10 = (_Float16)f10;
      const _Float16 g01 = (_Float16)f01, g11 = (_Float16)f11;
      const h4 s00 = {g00, g00, g00, g00};
      const h4 s10 = {g10, g10, g10, g10};
      const h4 s01 = {g01, g01, g01, g01};
      const h4 s11 = {g11, g11, g11, g11};
      const h4 acc = s00 * t00 + s10 * t10 + s01 * t01 + s11 * t11;  // v_pk_fma_f16
      *(h4*)(sAw + q * C + co) = acc;
    }
    wave_lds_order();   // same-wave DS in-order; just pin compiler schedule

    // --- 16x16x128 as four 16x16x32 WMMA steps; diagonal = correlations ---
    v8f cacc = {};
#pragma unroll
    for (int kk = 0; kk < 4; ++kk) {
      const v16h afrag = fragA_from_lds(sAw + (size_t)p * C, kk * 32, lane);
      cacc = __builtin_amdgcn_wmma_f32_16x16x32_f16(false, afrag, false, bfrag[kk],
                                                    (short)0, cacc, false, false);
    }
    wave_lds_order();   // next iteration's stores must not pass these loads

    // diagonal (j,j): j<8 held by lane j (v[j]); j>=8 by lane j+16 (v[j-8])
    if ((lane >> 4) == (p >> 3)) {
      const float val = cacc[p & 7] * 0.08838834764831845f;  // 1/sqrt(128)
      out[(((size_t)n * D + dep) * H + h) * W + (w0 + p)] = val;
    }
  }
}

extern "C" void kernel_launch(void* const* d_in, const int* in_sizes, int n_in,
                              void* d_out, int out_size, void* d_ws, size_t ws_size,
                              hipStream_t stream) {
  (void)in_sizes; (void)n_in; (void)out_size; (void)ws_size;
  const float* features = (const float*)d_in[0];
  const float* extr     = (const float*)d_in[1];
  const float* nearv    = (const float*)d_in[2];
  const float* farv     = (const float*)d_in[3];
  _Float16* T = (_Float16*)d_ws;
  float* P = (float*)((char*)d_ws + T_BYTES);
  float* out = (float*)d_out;

  k_transpose<<<dim3(H, NVIEW), 256, 0, stream>>>(features, T);
  k_setup<<<1, 64, 0, stream>>>(extr, nearv, farv, P);
  k_corr<<<dim3(W / 16, H, NVIEW), 256, 0, stream>>>(T, P, out);
}
